// CartesianPlaneEmbeddingNetwork_4406636446006
// MI455X (gfx1250) — matrix operand order repaired and checked by reading, not compiled, not run on hardware
//
#include <hip/hip_runtime.h>
#include <hip/hip_bf16.h>

typedef __attribute__((ext_vector_type(16))) _Float16 v16h;
typedef __attribute__((ext_vector_type(8)))  _Float16 v8h;
typedef __attribute__((ext_vector_type(8)))  float    v8f;

#define CDIM   128
#define WSTR   136      // LDS row stride in halves (bank-conflict padding)
#define W0F    30.0f

__launch_bounds__(256, 2)
__global__ void triplane_siren_fused(const float* __restrict__ coords,
                                     const float* __restrict__ plane,   // (128,256,256)
                                     const float* __restrict__ W1, const float* __restrict__ b1,
                                     const float* __restrict__ W2, const float* __restrict__ b2,
                                     const float* __restrict__ W3, const float* __restrict__ b3,
                                     float* __restrict__ out, int N) {
    __shared__ __align__(16) _Float16 w1s[CDIM * WSTR];
    __shared__ __align__(16) _Float16 w2s[CDIM * WSTR];
    __shared__ __align__(16) _Float16 hb[8][16 * WSTR];

    // ---- stage weights (f32 -> f16) into LDS, reused by all 8 waves ----
    for (int i = threadIdx.x; i < CDIM * CDIM; i += 256) {
        int n = i >> 7, k = i & 127;
        w1s[n * WSTR + k] = (_Float16)W1[i];
        w2s[n * WSTR + k] = (_Float16)W2[i];
    }
    __syncthreads();

    const int lane = threadIdx.x & 31;
    const int wave = threadIdx.x >> 5;
    const int hi   = lane >> 4;        // half-wave select (K-split for A/B frags)
    const int col  = lane & 15;        // A-row / B-column / D-column

    const int base = blockIdx.x * 128 + wave * 16;
    const int gp   = base + col;                  // point this lane gathers (A row = col)
    const int gpc  = (gp < N) ? gp : (N - 1);

    // ---- bilinear setup (align_corners=True, 256x256, shared x for xy & xz) ----
    const float gx = coords[gpc * 3 + 0];
    const float gy = coords[gpc * 3 + 1];
    const float gz = coords[gpc * 3 + 2];
    const float ix = (gx + 1.0f) * 127.5f;
    const float iy = (gy + 1.0f) * 127.5f;
    const float iz = (gz + 1.0f) * 127.5f;
    const float fx0 = floorf(ix), fy0 = floorf(iy), fz0 = floorf(iz);
    float wx1 = ix - fx0, wx0 = 1.0f - wx1;
    float wy1 = iy - fy0, wy0 = 1.0f - wy1;
    float wz1 = iz - fz0, wz0 = 1.0f - wz1;
    int xi0 = (int)fx0, yi0 = (int)fy0, zi0 = (int)fz0;
    xi0 = max(0, min(255, xi0)); yi0 = max(0, min(255, yi0)); zi0 = max(0, min(255, zi0));
    const int xi1 = min(255, xi0 + 1), yi1 = min(255, yi0 + 1), zi1 = min(255, zi0 + 1);
    const int oy0 = yi0 * 256, oy1 = yi1 * 256, oz0 = zi0 * 256, oz1 = zi1 * 256;

    // ---- gather features straight into WMMA A-fragment layout ----
    // A 16x32 as this code packs it (B uses same K order): lane&15 = row M; lanes<16 hold K: {0..15}, lanes>=16: +16.
    v16h aF[4];
#pragma unroll
    for (int kt = 0; kt < 4; ++kt) {
#pragma unroll
        for (int e = 0; e < 16; ++e) {
            const int off = e;                          // {0..15}
            const int c   = kt * 32 + hi * 16 + off;
            const float* pc = plane + (size_t)c * 65536;
            const float v00 = pc[oy0 + xi0], v01 = pc[oy0 + xi1];
            const float v10 = pc[oy1 + xi0], v11 = pc[oy1 + xi1];
            const float sxy = wy0 * (wx0 * v00 + wx1 * v01) + wy1 * (wx0 * v10 + wx1 * v11);
            const float u00 = pc[oz0 + xi0], u01 = pc[oz0 + xi1];
            const float u10 = pc[oz1 + xi0], u11 = pc[oz1 + xi1];
            const float sxz = wz0 * (wx0 * u00 + wx1 * u01) + wz1 * (wx0 * u10 + wx1 * u11);
            aF[kt][e] = (_Float16)(sxy * sxz * sxz);
        }
    }

    v8f acc[8];

    // B fragment address helper: W LDS row = output channel n, 16 contiguous halves of K
    auto bfrag_ptr = [&](const _Float16* ws, int i) -> const v8h* {
        const int kt = i >> 3, nt = i & 7;
        return (const v8h*)&ws[(nt * 16 + col) * WSTR + kt * 32 + hi * 16];
    };

    // ================= layer 1: feat @ W1^T (pipelined B-frag loads) =================
#pragma unroll
    for (int nt = 0; nt < 8; ++nt) { v8f z = {}; acc[nt] = z; }
    {
        const v8h* p0 = bfrag_ptr(w1s, 0);
        v8h blo = p0[0], bhi = p0[1];
#pragma unroll
        for (int i = 0; i < 32; ++i) {
            v8h nlo = blo, nhi = bhi;
            if (i < 31) {                       // issue next frag load before this WMMA
                const v8h* pn = bfrag_ptr(w1s, i + 1);
                nlo = pn[0]; nhi = pn[1];
            }
            const v16h bf = __builtin_shufflevector(blo, bhi,
                0,1,2,3,4,5,6,7,8,9,10,11,12,13,14,15);
            const int kt = i >> 3, nt = i & 7;
            acc[nt] = __builtin_amdgcn_wmma_f32_16x16x32_f16(
                false, aF[kt], false, bf, (short)0, acc[nt], false, false);
            blo = nlo; bhi = nhi;
        }
    }

    // bias + sin, bounce through per-wave LDS to rebuild A-fragments
#pragma unroll
    for (int nt = 0; nt < 8; ++nt) {
        const float bb = b1[nt * 16 + col];
#pragma unroll
        for (int r = 0; r < 8; ++r) {
            const float h = __sinf(W0F * (acc[nt][r] + bb));
            hb[wave][(r + hi * 8) * WSTR + nt * 16 + col] = (_Float16)h;
        }
    }
    asm volatile("s_wait_dscnt 0" ::: "memory");   // DS in-order within wave
#pragma unroll
    for (int kt = 0; kt < 4; ++kt) {
        const v8h* ap = (const v8h*)&hb[wave][col * WSTR + kt * 32 + hi * 16];
        const v8h a0 = ap[0], a1 = ap[1];          // K {0..7} and {8..15} (+hi*16)
        aF[kt] = __builtin_shufflevector(a0, a1,
            0,1,2,3,4,5,6,7,8,9,10,11,12,13,14,15);
    }

    // ================= layer 2: h @ W2^T (pipelined B-frag loads) =================
#pragma unroll
    for (int nt = 0; nt < 8; ++nt) { v8f z = {}; acc[nt] = z; }
    {
        const v8h* p0 = bfrag_ptr(w2s, 0);
        v8h blo = p0[0], bhi = p0[1];
#pragma unroll
        for (int i = 0; i < 32; ++i) {
            v8h nlo = blo, nhi = bhi;
            if (i < 31) {
                const v8h* pn = bfrag_ptr(w2s, i + 1);
                nlo = pn[0]; nhi = pn[1];
            }
            const v16h bf = __builtin_shufflevector(blo, bhi,
                0,1,2,3,4,5,6,7,8,9,10,11,12,13,14,15);
            const int kt = i >> 3, nt = i & 7;
            acc[nt] = __builtin_amdgcn_wmma_f32_16x16x32_f16(
                false, aF[kt], false, bf, (short)0, acc[nt], false, false);
            blo = nlo; bhi = nhi;
        }
    }

    // ================= layer 3: sin-activate, dot with W3, lane butterfly =================
    float part[8];
#pragma unroll
    for (int r = 0; r < 8; ++r) part[r] = 0.0f;
#pragma unroll
    for (int nt = 0; nt < 8; ++nt) {
        const float bb  = b2[nt * 16 + col];
        const float w3v = W3[nt * 16 + col];
#pragma unroll
        for (int r = 0; r < 8; ++r)
            part[r] += __sinf(W0F * (acc[nt][r] + bb)) * w3v;
    }
#pragma unroll
    for (int m = 1; m < 16; m <<= 1) {
#pragma unroll
        for (int r = 0; r < 8; ++r)
            part[r] += __shfl_xor(part[r], m, 32);
    }
    if (col == 0) {
        const float b3v = b3[0];
#pragma unroll
        for (int r = 0; r < 8; ++r) {
            const int g = base + r + hi * 8;      // lane0 -> M 0..7, lane16 -> M 8..15
            if (g < N) out[g] = part[r] + b3v;
        }
    }
}

extern "C" void kernel_launch(void* const* d_in, const int* in_sizes, int n_in,
                              void* d_out, int out_size, void* d_ws, size_t ws_size,
                              hipStream_t stream) {
    const float* coords = (const float*)d_in[0];   // (1,N,3)
    const float* plane  = (const float*)d_in[1];   // plane_xy (only plane used — bug preserved)
    // d_in[2] plane_yz, d_in[3] plane_xz: unused by the reference computation
    const float* W1 = (const float*)d_in[4];
    const float* b1 = (const float*)d_in[5];
    const float* W2 = (const float*)d_in[6];
    const float* b2 = (const float*)d_in[7];
    const float* W3 = (const float*)d_in[8];
    const float* b3 = (const float*)d_in[9];
    float* out = (float*)d_out;

    const int N = in_sizes[0] / 3;
    const int blocks = (N + 127) / 128;            // 128 points per block (8 waves x 16)
    triplane_siren_fused<<<blocks, 256, 0, stream>>>(coords, plane, W1, b1, W2, b2, W3, b3, out, N);
}
